// Transformer_48713519071247
// MI455X (gfx1250) — matrix operand order repaired and checked by reading, not compiled, not run on hardware
//
#include <hip/hip_runtime.h>
#include <hip/hip_bf16.h>

// ---------------- types ----------------
typedef __attribute__((ext_vector_type(16))) __bf16 v16bf;
typedef __attribute__((ext_vector_type(8)))  __bf16 v8bf;
typedef __attribute__((ext_vector_type(8)))  float  v8f;

// ---------------- config ----------------
#define NL   6
#define Hh   16
#define Dd   1024
#define DKk  64
#define Vv   32000
#define Bb   2
#define Ss   1024
#define DFFf 4096
#define BSr  (Bb * Ss)   // 2048 token rows

// ---------------- CDNA5 async-to-LDS helpers ----------------
// Raw LDS byte offset of a __shared__ object (addrspace(3) pointer value).
__device__ inline unsigned lds_off(const void* p) {
    return (unsigned)(uintptr_t)(__attribute__((address_space(3))) const void*)p;
}
// GLOBAL_LOAD_ASYNC_TO_LDS_B128: per-lane 16B global -> LDS, tracked by ASYNCcnt.
__device__ inline void async_copy_b128(unsigned lds_addr, const void* gaddr) {
    asm volatile("global_load_async_to_lds_b128 %0, %1, off"
                 :: "v"(lds_addr), "v"((unsigned long long)(uintptr_t)gaddr)
                 : "memory");
}
__device__ inline void wait_asynccnt0() {
    asm volatile("s_wait_asynccnt 0x0" ::: "memory");
}

// =====================================================================
// Weight fp32 -> bf16 transpose (WT[c][r] = (bf16)W[r][c]), batched in z
// =====================================================================
__global__ __launch_bounds__(256)
void transpose_bf16_kernel(const float* __restrict__ W, __bf16* __restrict__ WT,
                           int R, int C)
{
    long batch = blockIdx.z;
    W  += batch * (long)R * C;
    WT += batch * (long)R * C;
    __shared__ float tile[32][33];
    int r0 = blockIdx.y * 32, c0 = blockIdx.x * 32;
    int tx = threadIdx.x & 31, ty = threadIdx.x >> 5;
    #pragma unroll
    for (int i = ty; i < 32; i += 8) {
        int r = r0 + i, c = c0 + tx;
        tile[i][tx] = (r < R && c < C) ? W[(long)r * C + c] : 0.f;
    }
    __syncthreads();
    #pragma unroll
    for (int i = ty; i < 32; i += 8) {
        int c = c0 + i, r = r0 + tx;
        if (c < C && r < R) WT[(long)c * R + r] = (__bf16)tile[tx][i];
    }
}

// =====================================================================
// Embedding gather + positional encoding (base 1000, sin even / cos odd)
// =====================================================================
__global__ __launch_bounds__(256)
void embed_pos_kernel(const int* __restrict__ outputs, const float* __restrict__ embed,
                      float* __restrict__ x, __bf16* __restrict__ xb)
{
    int row = blockIdx.x;            // b*S + s
    int s   = row & (Ss - 1);
    int tok = outputs[row];
    const float LN1000 = 6.9077552789821368f;
    for (int d = threadIdx.x; d < Dd; d += 256) {
        float angle = (float)s * __expf(-((float)d / (float)Dd) * LN1000);
        float pe    = (d & 1) ? __cosf(angle) : __sinf(angle);
        float v     = embed[(long)tok * Dd + d] + pe;
        x [(long)row * Dd + d] = v;
        xb[(long)row * Dd + d] = (__bf16)v;
    }
}

// =====================================================================
// WMMA bf16 GEMM: C[m,n] = sum_k A[m,k] * BT[n,k]
// Block = 8 waves (256 thr), block tile 256(M) x 64(N); each wave owns
// 32x64 (8 f32 accumulators = 64 VGPRs).  B tile (64 n-rows x 64 k, 8KB,
// row-padded to 144B against LDS bank conflicts) is staged in LDS with
// GLOBAL_LOAD_ASYNC_TO_LDS_B128, double-buffered: one ASYNCcnt wait +
// one split workgroup barrier covers 16 WMMAs.  A fragments stream
// straight from global (two contiguous 16B chunks per lane).
// MODE: 0=f32 (+opt bias), 1=bf16, 2=bf16+relu, 3=bf16 transposed store
// Batched over blockIdx.z with z1 = z/zdiv, z2 = z%zdiv pointer offsets.
// Requires: M % 256 == 0, N % 64 == 0, K % 64 == 0 (all call sites comply).
// =====================================================================
#define BROW 72                    // padded B-tile row stride (elements)
#define BBUF (64 * BROW)           // elements per buffer (9216 B)

template <int MODE>
__global__ __launch_bounds__(256)
void wmma_gemm(const __bf16* __restrict__ A, const __bf16* __restrict__ BT,
               void* __restrict__ Cv, const float* __restrict__ bias,
               int M, int N, int K, int lda, int ldbt, int ldc,
               int zdiv, long sA1, long sA2, long sB1, long sB2, long sC1, long sC2)
{
    __shared__ __attribute__((aligned(128))) __bf16 btile[2][BBUF];

    int z = blockIdx.z, z1 = z / zdiv, z2 = z % zdiv;
    A  += z1 * sA1 + z2 * sA2;
    BT += z1 * sB1 + z2 * sB2;
    long cbase = z1 * sC1 + z2 * sC2;

    int tid   = threadIdx.x;
    int wave  = tid >> 5;
    int lane  = tid & 31;
    int lhalf = lane >> 4, l16 = lane & 15;
    int m0 = (blockIdx.y * 8 + wave) * 32;   // 32 rows per wave
    int n0 = blockIdx.x * 64;

    // Cooperative B staging: 64 rows x 64 k bf16 = 512 x 16B chunks,
    // 2 chunks/thread: row = tid>>2, chunk cols (tid&3)*2 + {0,1}.
    int srow = tid >> 2;
    int scol = (tid & 3) * 2;
    const __bf16* gsrc = BT + (long)(n0 + srow) * ldbt + scol * 8;
    unsigned lbase = lds_off(&btile[0][0]);
    unsigned lrow  = (unsigned)(srow * (BROW * 2) + scol * 16);

    v8f zero = {0.f, 0.f, 0.f, 0.f, 0.f, 0.f, 0.f, 0.f};
    v8f acc[2][4];
    #pragma unroll
    for (int mi = 0; mi < 2; ++mi)
        #pragma unroll
        for (int t = 0; t < 4; ++t) acc[mi][t] = zero;

    auto stage = [&](int buf, int k0) {
        unsigned lb = lbase + (unsigned)buf * (BBUF * 2) + lrow;
        const __bf16* g = gsrc + k0;
        async_copy_b128(lb,      g);
        async_copy_b128(lb + 16, g + 8);
    };

    stage(0, 0);                       // prologue: first B tile -> buffer 0
    #pragma unroll 2
    for (int k0 = 0; k0 < K; k0 += 64) {
        int buf = (k0 >> 6) & 1;
        wait_asynccnt0();              // this wave's async chunks landed
        __syncthreads();               // tile visible; prev buffer free
        if (k0 + 64 < K) stage(buf ^ 1, k0 + 64);
        #pragma unroll
        for (int kk = 0; kk < 2; ++kk) {
            // A fragments for the wave's two 16-row tiles
            v16bf afrag[2];
            #pragma unroll
            for (int mi = 0; mi < 2; ++mi) {
                const __bf16* pa = A + (long)(m0 + mi * 16 + l16) * lda
                                     + k0 + kk * 32 + lhalf * 8;
                v8bf alo = *(const v8bf*)pa;
                v8bf ahi = *(const v8bf*)(pa + 16);
                #pragma unroll
                for (int i = 0; i < 8; ++i) { afrag[mi][i] = alo[i]; afrag[mi][i + 8] = ahi[i]; }
            }
            // B fragments from LDS, 8 WMMAs per kk
            #pragma unroll
            for (int t = 0; t < 4; ++t) {
                const __bf16* pb = &btile[buf][(t * 16 + l16) * BROW + kk * 32 + lhalf * 8];
                v8bf blo = *(const v8bf*)pb;
                v8bf bhi = *(const v8bf*)(pb + 16);
                v16bf bfrag;
                #pragma unroll
                for (int i = 0; i < 8; ++i) { bfrag[i] = blo[i]; bfrag[i + 8] = bhi[i]; }
                #pragma unroll
                for (int mi = 0; mi < 2; ++mi)
                    acc[mi][t] = __builtin_amdgcn_wmma_f32_16x16x32_bf16(
                        false, afrag[mi], false, bfrag, (short)0, acc[mi][t], false, false);
            }
        }
    }

    #pragma unroll
    for (int mi = 0; mi < 2; ++mi)
        #pragma unroll
        for (int t = 0; t < 4; ++t) {
            int n = n0 + t * 16 + l16;
            #pragma unroll
            for (int r = 0; r < 8; ++r) {
                int m = m0 + mi * 16 + r + lhalf * 8;
                float val = acc[mi][t][r];
                if (MODE == 0) {
                    float* c = (float*)Cv;
                    float bb = bias ? bias[n] : 0.f;
                    c[cbase + (long)m * ldc + n] = val + bb;
                } else if (MODE == 1) {
                    ((__bf16*)Cv)[cbase + (long)m * ldc + n] = (__bf16)val;
                } else if (MODE == 2) {
                    ((__bf16*)Cv)[cbase + (long)m * ldc + n] = (__bf16)(val > 0.f ? val : 0.f);
                } else {  // transposed bf16 store: C[n][m]
                    ((__bf16*)Cv)[cbase + (long)n * ldc + m] = (__bf16)val;
                }
            }
        }
}

// =====================================================================
// Fused causal attention (flash-style online softmax), one wave per
// 16-row Q tile, key blocks of 32.  q,k: [B,H,S,DK] bf16; vT: [B,H,DK,S]
// bf16; out: fp32 [B*S, H*DK] (concat-of-heads, ready for residual+LN).
// =====================================================================
__global__ __launch_bounds__(128)
void attn_kernel(const __bf16* __restrict__ q, const __bf16* __restrict__ k,
                 const __bf16* __restrict__ vT, float* __restrict__ out)
{
    __shared__ __attribute__((aligned(64))) __bf16 plds[4][16 * 32];

    int z = blockIdx.z, b = z / Hh, h = z % Hh;
    int wave  = threadIdx.x >> 5;
    int lane  = threadIdx.x & 31;
    int lhalf = lane >> 4, l16 = lane & 15;
    int q0 = blockIdx.x * 64 + wave * 16;

    const __bf16* qb = q  + (long)z * Ss * DKk;
    const __bf16* kb = k  + (long)z * Ss * DKk;
    const __bf16* vb = vT + (long)z * DKk * Ss;

    v16bf aq[2];
    #pragma unroll
    for (int kk = 0; kk < 2; ++kk) {
        const __bf16* pa = qb + (long)(q0 + l16) * DKk + kk * 32 + lhalf * 16;
        v8bf lo = *(const v8bf*)pa;
        v8bf hi = *(const v8bf*)(pa + 8);
        #pragma unroll
        for (int i = 0; i < 8; ++i) { aq[kk][i] = lo[i]; aq[kk][i + 8] = hi[i]; }
    }

    v8f zero = {0.f, 0.f, 0.f, 0.f, 0.f, 0.f, 0.f, 0.f};
    v8f o[4];
    #pragma unroll
    for (int t = 0; t < 4; ++t) o[t] = zero;
    float mrow[8], lrow[8];
    #pragma unroll
    for (int r = 0; r < 8; ++r) { mrow[r] = -3.0e38f; lrow[r] = 0.f; }

    const float sc = 0.125f;  // 1/sqrt(DK)
    int kmax = q0 + 16;
    for (int key0 = 0; key0 < kmax; key0 += 32) {
        v8f s2[2];
        #pragma unroll
        for (int cb = 0; cb < 2; ++cb) {
            s2[cb] = zero;
            #pragma unroll
            for (int kk = 0; kk < 2; ++kk) {
                const __bf16* pb = kb + (long)(key0 + cb * 16 + l16) * DKk + kk * 32 + lhalf * 16;
                v16bf bf = *(const v16bf*)pb;
                s2[cb] = __builtin_amdgcn_wmma_f32_16x16x32_bf16(
                    false, aq[kk], false, bf, (short)0, s2[cb], false, false);
            }
        }
        float bmax[8];
        #pragma unroll
        for (int r = 0; r < 8; ++r) {
            int row = q0 + r + lhalf * 8;
            int c0 = key0 + l16, c1 = key0 + 16 + l16;
            float v0 = s2[0][r] * sc, v1 = s2[1][r] * sc;
            v0 = (c0 <= row && v0 != 0.f) ? v0 : -3.0e38f;
            v1 = (c1 <= row && v1 != 0.f) ? v1 : -3.0e38f;
            s2[0][r] = v0; s2[1][r] = v1;
            bmax[r] = fmaxf(v0, v1);
        }
        #pragma unroll
        for (int off = 1; off < 16; off <<= 1)
            #pragma unroll
            for (int r = 0; r < 8; ++r)
                bmax[r] = fmaxf(bmax[r], __shfl_xor(bmax[r], off, 32));
        float scalef[8], psum[8];
        #pragma unroll
        for (int r = 0; r < 8; ++r) {
            float mnew = fmaxf(mrow[r], bmax[r]);
            scalef[r]  = __expf(mrow[r] - mnew);
            mrow[r]    = mnew;
            float p0 = __expf(s2[0][r] - mnew);
            float p1 = __expf(s2[1][r] - mnew);
            s2[0][r] = p0; s2[1][r] = p1;
            psum[r]  = p0 + p1;
        }
        #pragma unroll
        for (int off = 1; off < 16; off <<= 1)
            #pragma unroll
            for (int r = 0; r < 8; ++r)
                psum[r] += __shfl_xor(psum[r], off, 32);
        #pragma unroll
        for (int r = 0; r < 8; ++r) {
            lrow[r] = lrow[r] * scalef[r] + psum[r];
            #pragma unroll
            for (int t = 0; t < 4; ++t) o[t][r] *= scalef[r];
        }
        // P: C layout -> A layout via per-wave LDS tile
        __bf16* pl = plds[wave];
        #pragma unroll
        for (int cb = 0; cb < 2; ++cb)
            #pragma unroll
            for (int r = 0; r < 8; ++r)
                pl[(r + lhalf * 8) * 32 + cb * 16 + l16] = (__bf16)s2[cb][r];
        const __bf16* par = pl + l16 * 32;
        v8bf plo = *(const v8bf*)(par + lhalf * 16);
        v8bf phi = *(const v8bf*)(par + 8 + lhalf * 16);
        v16bf pfrag;
        #pragma unroll
        for (int i = 0; i < 8; ++i) { pfrag[i] = plo[i]; pfrag[i + 8] = phi[i]; }
        #pragma unroll
        for (int t = 0; t < 4; ++t) {
            const __bf16* pb = vb + (long)(t * 16 + l16) * Ss + key0 + lhalf * 16;
            v16bf bf = *(const v16bf*)pb;
            o[t] = __builtin_amdgcn_wmma_f32_16x16x32_bf16(
                false, pfrag, false, bf, (short)0, o[t], false, false);
        }
    }
    float inv[8];
    #pragma unroll
    for (int r = 0; r < 8; ++r) inv[r] = 1.f / lrow[r];
    #pragma unroll
    for (int t = 0; t < 4; ++t)
        #pragma unroll
        for (int r = 0; r < 8; ++r) {
            int row = q0 + r + lhalf * 8;
            int col = h * DKk + t * 16 + l16;
            out[((long)b * Ss + row) * Dd + col] = o[t][r] * inv[r];
        }
}

// =====================================================================
// residual add + LayerNorm: y = LN(a + res) * g + b
// =====================================================================
__global__ __launch_bounds__(256)
void add_ln_kernel(const float* __restrict__ a, const float* __restrict__ res,
                   const float* __restrict__ g, const float* __restrict__ bta,
                   float* __restrict__ xout, __bf16* __restrict__ xbout)
{
    long row = blockIdx.x;
    const float* ar = a   + row * Dd;
    const float* rr = res + row * Dd;
    __shared__ float red1[8], red2[8];
    int lane = threadIdx.x & 31, wv = threadIdx.x >> 5;

    float vals[4], s = 0.f;
    #pragma unroll
    for (int i = 0; i < 4; ++i) {
        float v = ar[threadIdx.x + i * 256] + rr[threadIdx.x + i * 256];
        vals[i] = v; s += v;
    }
    #pragma unroll
    for (int off = 16; off; off >>= 1) s += __shfl_xor(s, off, 32);
    if (!lane) red1[wv] = s;
    __syncthreads();
    float tot = 0.f;
    #pragma unroll
    for (int i = 0; i < 8; ++i) tot += red1[i];
    float mean = tot * (1.f / Dd);

    float vs = 0.f;
    #pragma unroll
    for (int i = 0; i < 4; ++i) { float d = vals[i] - mean; vs += d * d; }
    #pragma unroll
    for (int off = 16; off; off >>= 1) vs += __shfl_xor(vs, off, 32);
    if (!lane) red2[wv] = vs;
    __syncthreads();
    float tot2 = 0.f;
    #pragma unroll
    for (int i = 0; i < 8; ++i) tot2 += red2[i];
    float rstd = rsqrtf(tot2 * (1.f / Dd) + 1e-5f);

    #pragma unroll
    for (int i = 0; i < 4; ++i) {
        int d0 = threadIdx.x + i * 256;
        float y = (vals[i] - mean) * rstd * g[d0] + bta[d0];
        xout [row * Dd + d0] = y;
        xbout[row * Dd + d0] = (__bf16)y;
    }
}

// =====================================================================
// launcher
// =====================================================================
extern "C" void kernel_launch(void* const* d_in, const int* in_sizes, int n_in,
                              void* d_out, int out_size, void* d_ws, size_t ws_size,
                              hipStream_t stream)
{
    const int*   outputs = (const int*)  d_in[0];
    const float* embed   = (const float*)d_in[1];
    const float* Wq      = (const float*)d_in[2];
    const float* Wk      = (const float*)d_in[3];
    const float* Wv      = (const float*)d_in[4];
    const float* W1      = (const float*)d_in[5];
    const float* W2      = (const float*)d_in[6];
    const float* ln1_g   = (const float*)d_in[7];
    const float* ln1_b   = (const float*)d_in[8];
    const float* ln2_g   = (const float*)d_in[9];
    const float* ln2_b   = (const float*)d_in[10];
    const float* Wout    = (const float*)d_in[11];
    const float* bout    = (const float*)d_in[12];
    float* logits = (float*)d_out;

    char* ws = (char*)d_ws;
    size_t off = 0;
    auto carve = [&](size_t bytes) -> char* {
        char* p = ws + off;
        off += (bytes + 255) & ~(size_t)255;
        return p;
    };
    __bf16* WqT   = (__bf16*)carve((size_t)NL * Hh * DKk * Dd * 2);
    __bf16* WkT   = (__bf16*)carve((size_t)NL * Hh * DKk * Dd * 2);
    __bf16* WvT   = (__bf16*)carve((size_t)NL * Hh * DKk * Dd * 2);
    __bf16* W1T   = (__bf16*)carve((size_t)NL * DFFf * Dd * 2);
    __bf16* W2T   = (__bf16*)carve((size_t)NL * Dd * DFFf * 2);
    __bf16* WoutT = (__bf16*)carve((size_t)Vv * Dd * 2);
    float*  x     = (float*) carve((size_t)BSr * Dd * 4);
    __bf16* xb    = (__bf16*)carve((size_t)BSr * Dd * 2);
    float*  tmpf  = (float*) carve((size_t)BSr * Dd * 4);
    __bf16* qbuf  = (__bf16*)carve((size_t)Bb * Hh * Ss * DKk * 2);
    __bf16* kbuf  = (__bf16*)carve((size_t)Bb * Hh * Ss * DKk * 2);
    __bf16* vTbuf = (__bf16*)carve((size_t)Bb * Hh * DKk * Ss * 2);
    __bf16* hb    = (__bf16*)carve((size_t)BSr * DFFf * 2);
    (void)in_sizes; (void)n_in; (void)out_size; (void)ws_size;

    // ---- weights -> bf16 transposed [N,K] ----
    transpose_bf16_kernel<<<dim3(DKk/32, Dd/32, NL*Hh), 256, 0, stream>>>(Wq, WqT, Dd, DKk);
    transpose_bf16_kernel<<<dim3(DKk/32, Dd/32, NL*Hh), 256, 0, stream>>>(Wk, WkT, Dd, DKk);
    transpose_bf16_kernel<<<dim3(DKk/32, Dd/32, NL*Hh), 256, 0, stream>>>(Wv, WvT, Dd, DKk);
    transpose_bf16_kernel<<<dim3(DFFf/32, Dd/32, NL), 256, 0, stream>>>(W1, W1T, Dd, DFFf);
    transpose_bf16_kernel<<<dim3(Dd/32, DFFf/32, NL), 256, 0, stream>>>(W2, W2T, DFFf, Dd);
    transpose_bf16_kernel<<<dim3(Vv/32, Dd/32, 1), 256, 0, stream>>>(Wout, WoutT, Dd, Vv);

    // ---- embedding + positional encoding ----
    embed_pos_kernel<<<BSr, 256, 0, stream>>>(outputs, embed, x, xb);

    const long sQC = (long)Hh * Ss * DKk;   // per-b stride in q/k [B,H,S,DK]
    const long sVC = (long)Hh * DKk * Ss;   // per-b stride in vT [B,H,DK,S]

    for (int l = 0; l < NL; ++l) {
        const __bf16* wq = WqT + (size_t)l * Hh * DKk * Dd;
        const __bf16* wk = WkT + (size_t)l * Hh * DKk * Dd;
        const __bf16* wv = WvT + (size_t)l * Hh * DKk * Dd;
        // Q/K: [B,H,S,DK] bf16; V: transposed store to [B,H,DK,S]
        wmma_gemm<1><<<dim3(1, Ss/256, Bb*Hh), 256, 0, stream>>>(
            xb, wq, qbuf, nullptr, Ss, DKk, Dd, Dd, Dd, DKk,
            Hh, (long)Ss*Dd, 0L, 0L, (long)DKk*Dd, sQC, (long)Ss*DKk);
        wmma_gemm<1><<<dim3(1, Ss/256, Bb*Hh), 256, 0, stream>>>(
            xb, wk, kbuf, nullptr, Ss, DKk, Dd, Dd, Dd, DKk,
            Hh, (long)Ss*Dd, 0L, 0L, (long)DKk*Dd, sQC, (long)Ss*DKk);
        wmma_gemm<3><<<dim3(1, Ss/256, Bb*Hh), 256, 0, stream>>>(
            xb, wv, vTbuf, nullptr, Ss, DKk, Dd, Dd, Dd, Ss,
            Hh, (long)Ss*Dd, 0L, 0L, (long)DKk*Dd, sVC, (long)DKk*Ss);
        // fused causal attention -> tmpf (concat heads, fp32)
        attn_kernel<<<dim3(Ss/64, 1, Bb*Hh), 128, 0, stream>>>(qbuf, kbuf, vTbuf, tmpf);
        // x = LN(x + attn)
        add_ln_kernel<<<BSr, 256, 0, stream>>>(tmpf, x, ln1_g + (size_t)l*Dd,
                                               ln1_b + (size_t)l*Dd, x, xb);
        // FFN
        wmma_gemm<2><<<dim3(DFFf/64, BSr/256, 1), 256, 0, stream>>>(
            xb, W1T + (size_t)l*DFFf*Dd, hb, nullptr,
            BSr, DFFf, Dd, Dd, Dd, DFFf, 1, 0L,0L,0L,0L,0L,0L);
        wmma_gemm<0><<<dim3(Dd/64, BSr/256, 1), 256, 0, stream>>>(
            hb, W2T + (size_t)l*Dd*DFFf, tmpf, nullptr,
            BSr, Dd, DFFf, DFFf, DFFf, Dd, 1, 0L,0L,0L,0L,0L,0L);
        // x = LN(x + ffn)
        add_ln_kernel<<<BSr, 256, 0, stream>>>(tmpf, x, ln2_g + (size_t)l*Dd,
                                               ln2_b + (size_t)l*Dd, x, xb);
    }
    // ---- output projection with bias -> logits fp32 ----
    wmma_gemm<0><<<dim3(Vv/64, BSr/256, 1), 256, 0, stream>>>(
        xb, WoutT, logits, bout, BSr, Vv, Dd, Dd, Dd, Vv, 1, 0L,0L,0L,0L,0L,0L);
}
